// AttentionOverride_20452634263842
// MI455X (gfx1250) — matrix operand order repaired and checked by reading, not compiled, not run on hardware
//
#include <hip/hip_runtime.h>
#include <hip/hip_bf16.h>

typedef _Float16 half_t;
typedef _Float16 v16h __attribute__((ext_vector_type(16)));
typedef _Float16 v8h  __attribute__((ext_vector_type(8)));
typedef float    v8f  __attribute__((ext_vector_type(8)));

#define WMMA_F16(a, b, c) \
  __builtin_amdgcn_wmma_f32_16x16x32_f16(false, (a), false, (b), (short)0, (c), false, false)

constexpr int Bb  = 4;
constexpr int Ss  = 1024;
constexpr int Dd  = 1024;
constexpr int Hh  = 16;
constexpr int HDd = 64;

// ---------------------------------------------------------------------------
// Kernel 0: convert x -> f16, w_attn -> f16 transposed [3D][D], w_proj -> f16
// transposed [D][D].
// ---------------------------------------------------------------------------
__global__ void cvt_kernel(const float* __restrict__ x,
                           const float* __restrict__ w_attn,
                           const float* __restrict__ w_proj,
                           half_t* __restrict__ x16,
                           half_t* __restrict__ wattnT,
                           half_t* __restrict__ wprojT) {
  long i = (long)blockIdx.x * blockDim.x + threadIdx.x;
  const long nx  = (long)Bb * Ss * Dd;   // 4M
  const long nwa = (long)Dd * 3 * Dd;    // 3M
  const long nwp = (long)Dd * Dd;        // 1M
  if (i < nx)  x16[i] = (half_t)x[i];
  if (i < nwa) {
    int k = (int)(i / (3 * Dd));
    int n = (int)(i % (3 * Dd));
    wattnT[(long)n * Dd + k] = (half_t)w_attn[i];
  }
  if (i < nwp) {
    int k = (int)(i / Dd);
    int n = (int)(i % Dd);
    wprojT[(long)n * Dd + k] = (half_t)w_proj[i];
  }
}

// ---------------------------------------------------------------------------
// Kernel 1: QKV GEMM.  C = x16[4096,1024] @ w_attn + b_attn  (N = 3072)
// Epilogue routes q/k/v into head-layout f16 buffers and writes f32 present.
// ---------------------------------------------------------------------------
__global__ __launch_bounds__(256, 2)
void qkv_gemm(const half_t* __restrict__ A, const half_t* __restrict__ Bt,
              const float* __restrict__ bias,
              half_t* __restrict__ q16, half_t* __restrict__ k16,
              half_t* __restrict__ vT16, float* __restrict__ present) {
  __shared__ half_t As[128][40];
  __shared__ half_t Bs[128][40];
  const int tid  = threadIdx.x;
  const int lane = tid & 31;
  const int wv   = tid >> 5;
  const int wm   = wv >> 2;      // 0..1
  const int wn   = wv & 3;       // 0..3
  const int m0   = blockIdx.x * 128;
  const int n0   = blockIdx.y * 128;
  const int ml   = lane & 15;
  const int kbA  = (lane >> 4) * 16;
  const int kbB  = (lane >> 4) * 16;
  const int rbase = (lane >> 4) * 8;

  v8f acc[4][2] = {};
  for (int k0 = 0; k0 < Dd; k0 += 32) {
    __syncthreads();
    {
      int r = tid >> 1, c = (tid & 1) * 16;
      const v8h* sa = (const v8h*)(A + (long)(m0 + r) * Dd + k0 + c);
      *(v8h*)&As[r][c]     = sa[0];
      *(v8h*)&As[r][c + 8] = sa[1];
      const v8h* sb = (const v8h*)(Bt + (long)(n0 + r) * Dd + k0 + c);
      *(v8h*)&Bs[r][c]     = sb[0];
      *(v8h*)&Bs[r][c + 8] = sb[1];
      if (k0 + 32 < Dd) {
        __builtin_prefetch(A + (long)(m0 + r) * Dd + k0 + 32 + c, 0, 1);
        __builtin_prefetch(Bt + (long)(n0 + r) * Dd + k0 + 32 + c, 0, 1);
      }
    }
    __syncthreads();

    v16h bf[2];
#pragma unroll
    for (int ni = 0; ni < 2; ++ni) {
      int n = wn * 32 + ni * 16 + ml;
      bf[ni] = *(const v16h*)&Bs[n][kbB];
    }
#pragma unroll
    for (int mi = 0; mi < 4; ++mi) {
      int m = wm * 64 + mi * 16 + ml;
      v8h lo = *(const v8h*)&As[m][kbA];
      v8h hi = *(const v8h*)&As[m][kbA + 8];
      v16h af = __builtin_shufflevector(lo, hi, 0, 1, 2, 3, 4, 5, 6, 7,
                                        8, 9, 10, 11, 12, 13, 14, 15);
      acc[mi][0] = WMMA_F16(af, bf[0], acc[mi][0]);
      acc[mi][1] = WMMA_F16(af, bf[1], acc[mi][1]);
    }
  }

  const int seg = n0 >> 10;  // 0=q, 1=k, 2=v (uniform per block: 1024 % 128 == 0)
#pragma unroll
  for (int mi = 0; mi < 4; ++mi) {
#pragma unroll
    for (int ni = 0; ni < 2; ++ni) {
      int col = n0 + wn * 32 + ni * 16 + ml;   // [0,3072)
      float bsv = bias[col];
      int c  = col & 1023;
      int h  = c >> 6;
      int hd = c & 63;
#pragma unroll
      for (int r = 0; r < 8; ++r) {
        int row = m0 + wm * 64 + mi * 16 + rbase + r;
        int b = row >> 10, s = row & 1023;
        float val = acc[mi][ni][r] + bsv;
        long bh = (long)b * Hh + h;
        if (seg == 0) {
          q16[(bh * Ss + s) * HDd + hd] = (half_t)val;
        } else if (seg == 1) {
          k16[(bh * Ss + s) * HDd + hd] = (half_t)val;
          present[(bh * Ss + s) * HDd + hd] = val;
        } else {
          vT16[(bh * HDd + hd) * Ss + s] = (half_t)val;
          present[(long)Bb * Hh * Ss * HDd + (bh * Ss + s) * HDd + hd] = val;
        }
      }
    }
  }
}

// ---------------------------------------------------------------------------
// Kernel 2: fused attention per (b,h).  4 waves/WG, 16 query rows per wave.
// Two-pass softmax (recompute scores) + post-softmax override, P@V via WMMA.
// ---------------------------------------------------------------------------
__global__ __launch_bounds__(128, 4)
void attn_kernel(const half_t* __restrict__ q16, const half_t* __restrict__ k16,
                 const half_t* __restrict__ vT16,
                 const float* __restrict__ ov, const unsigned char* __restrict__ msk,
                 half_t* __restrict__ a16) {
  __shared__ half_t Ps[4][16][40];
  const int tid   = threadIdx.x;
  const int lane  = tid & 31;
  const int wv    = tid >> 5;
  const int bh    = blockIdx.y;          // 0..63
  const int h     = bh & (Hh - 1);
  const int bat   = bh >> 4;
  const int q0    = blockIdx.x * 64 + wv * 16;
  const int ml    = lane & 15;
  const int kbA   = (lane >> 4) * 16;
  const int kbB   = (lane >> 4) * 16;
  const int rbase = (lane >> 4) * 8;
  const long base = (long)bh * Ss * HDd;

  // Load Q A-frags once, pre-scaled by 1/sqrt(64) = 0.125.
  v16h qf[2];
#pragma unroll
  for (int c = 0; c < 2; ++c) {
    const half_t* qr = q16 + base + (long)(q0 + ml) * HDd + c * 32;
    v8h lo = *(const v8h*)(qr + kbA);
    v8h hi = *(const v8h*)(qr + kbA + 8);
#pragma unroll
    for (int j = 0; j < 8; ++j) {
      qf[c][j]     = lo[j] * (half_t)0.125f;
      qf[c][j + 8] = hi[j] * (half_t)0.125f;
    }
  }

  float mrow[8], lrow[8];
#pragma unroll
  for (int r = 0; r < 8; ++r) { mrow[r] = -3.0e38f; lrow[r] = 0.0f; }

  // ---- Pass 1: per-lane online max / sum-exp over all keys ----
  for (int n0 = 0; n0 < Ss; n0 += 16) {
    v8f sf = {};
#pragma unroll
    for (int c = 0; c < 2; ++c) {
      v16h kf = *(const v16h*)(k16 + base + (long)(n0 + ml) * HDd + c * 32 + kbB);
      sf = WMMA_F16(qf[c], kf, sf);
    }
    int key = n0 + ml;
#pragma unroll
    for (int r = 0; r < 8; ++r) {
      int qrow = q0 + rbase + r;
      float s = (key <= qrow) ? sf[r] : -10000.0f;
      float mn = fmaxf(mrow[r], s);
      lrow[r] = lrow[r] * __expf(mrow[r] - mn) + __expf(s - mn);
      mrow[r] = mn;
    }
  }
  // Cross-lane reduce within each 16-lane group (rows 0-7 / 8-15 split).
#pragma unroll
  for (int mask = 1; mask < 16; mask <<= 1) {
#pragma unroll
    for (int r = 0; r < 8; ++r) {
      float mo = __shfl_xor(mrow[r], mask);
      float lo = __shfl_xor(lrow[r], mask);
      float mn = fmaxf(mrow[r], mo);
      lrow[r] = lrow[r] * __expf(mrow[r] - mn) + lo * __expf(mo - mn);
      mrow[r] = mn;
    }
  }
  float linv[8];
#pragma unroll
  for (int r = 0; r < 8; ++r) linv[r] = 1.0f / lrow[r];

  // ---- Pass 2: recompute scores, p = override?ov:exp(s-m)/l, O += P @ V ----
  v8f Oacc[4] = {};
  for (int t0 = 0; t0 < Ss; t0 += 32) {
#pragma unroll
    for (int sb = 0; sb < 2; ++sb) {
      int n0 = t0 + sb * 16;
      v8f sf = {};
#pragma unroll
      for (int c = 0; c < 2; ++c) {
        v16h kf = *(const v16h*)(k16 + base + (long)(n0 + ml) * HDd + c * 32 + kbB);
        sf = WMMA_F16(qf[c], kf, sf);
      }
      int key = n0 + ml;
#pragma unroll
      for (int r = 0; r < 8; ++r) {
        int qrow = q0 + rbase + r;
        float s = (key <= qrow) ? sf[r] : -10000.0f;
        float p = __expf(s - mrow[r]) * linv[r];
        long oidx = (long)h * Ss * Ss + (long)qrow * Ss + key;
        p = msk[oidx] ? ov[oidx] : p;  // post-softmax override
        Ps[wv][rbase + r][sb * 16 + ml] = (half_t)p;
      }
    }
    __syncthreads();
    v8h lo = *(const v8h*)&Ps[wv][ml][kbA];
    v8h hi = *(const v8h*)&Ps[wv][ml][kbA + 8];
    v16h pf = __builtin_shufflevector(lo, hi, 0, 1, 2, 3, 4, 5, 6, 7,
                                      8, 9, 10, 11, 12, 13, 14, 15);
#pragma unroll
    for (int nb = 0; nb < 4; ++nb) {
      int hd = nb * 16 + ml;
      v16h vf = *(const v16h*)(vT16 + (long)bh * HDd * Ss + (long)hd * Ss + t0 + kbB);
      Oacc[nb] = WMMA_F16(pf, vf, Oacc[nb]);
    }
    __syncthreads();
  }

  // Write merged-head output rows into a16 [B*S, D].
#pragma unroll
  for (int nb = 0; nb < 4; ++nb) {
    int col = h * HDd + nb * 16 + ml;
#pragma unroll
    for (int r = 0; r < 8; ++r) {
      int qrow = q0 + rbase + r;
      a16[((long)bat * Ss + qrow) * Dd + col] = (half_t)Oacc[nb][r];
    }
  }
}

// ---------------------------------------------------------------------------
// Kernel 3: projection GEMM.  out = a16[4096,1024] @ w_proj + b_proj (f32)
// ---------------------------------------------------------------------------
__global__ __launch_bounds__(256, 2)
void proj_gemm(const half_t* __restrict__ A, const half_t* __restrict__ Bt,
               const float* __restrict__ bias, float* __restrict__ out) {
  __shared__ half_t As[128][40];
  __shared__ half_t Bs[128][40];
  const int tid  = threadIdx.x;
  const int lane = tid & 31;
  const int wv   = tid >> 5;
  const int wm   = wv >> 2;
  const int wn   = wv & 3;
  const int m0   = blockIdx.x * 128;
  const int n0   = blockIdx.y * 128;
  const int ml   = lane & 15;
  const int kbA  = (lane >> 4) * 16;
  const int kbB  = (lane >> 4) * 16;
  const int rbase = (lane >> 4) * 8;

  v8f acc[4][2] = {};
  for (int k0 = 0; k0 < Dd; k0 += 32) {
    __syncthreads();
    {
      int r = tid >> 1, c = (tid & 1) * 16;
      const v8h* sa = (const v8h*)(A + (long)(m0 + r) * Dd + k0 + c);
      *(v8h*)&As[r][c]     = sa[0];
      *(v8h*)&As[r][c + 8] = sa[1];
      const v8h* sb = (const v8h*)(Bt + (long)(n0 + r) * Dd + k0 + c);
      *(v8h*)&Bs[r][c]     = sb[0];
      *(v8h*)&Bs[r][c + 8] = sb[1];
    }
    __syncthreads();

    v16h bf[2];
#pragma unroll
    for (int ni = 0; ni < 2; ++ni) {
      int n = wn * 32 + ni * 16 + ml;
      bf[ni] = *(const v16h*)&Bs[n][kbB];
    }
#pragma unroll
    for (int mi = 0; mi < 4; ++mi) {
      int m = wm * 64 + mi * 16 + ml;
      v8h lo = *(const v8h*)&As[m][kbA];
      v8h hi = *(const v8h*)&As[m][kbA + 8];
      v16h af = __builtin_shufflevector(lo, hi, 0, 1, 2, 3, 4, 5, 6, 7,
                                        8, 9, 10, 11, 12, 13, 14, 15);
      acc[mi][0] = WMMA_F16(af, bf[0], acc[mi][0]);
      acc[mi][1] = WMMA_F16(af, bf[1], acc[mi][1]);
    }
  }

#pragma unroll
  for (int mi = 0; mi < 4; ++mi) {
#pragma unroll
    for (int ni = 0; ni < 2; ++ni) {
      int col = n0 + wn * 32 + ni * 16 + ml;
      float bsv = bias[col];
#pragma unroll
      for (int r = 0; r < 8; ++r) {
        int row = m0 + wm * 64 + mi * 16 + rbase + r;
        out[(long)row * Dd + col] = acc[mi][ni][r] + bsv;
      }
    }
  }
}

// ---------------------------------------------------------------------------
extern "C" void kernel_launch(void* const* d_in, const int* in_sizes, int n_in,
                              void* d_out, int out_size, void* d_ws, size_t ws_size,
                              hipStream_t stream) {
  const float* x       = (const float*)d_in[0];
  const float* w_attn  = (const float*)d_in[1];
  const float* b_attn  = (const float*)d_in[2];
  const float* w_proj  = (const float*)d_in[3];
  const float* b_proj  = (const float*)d_in[4];
  const float* ov      = (const float*)d_in[5];
  const unsigned char* msk = (const unsigned char*)d_in[6];

  float* out = (float*)d_out;              // [a (4M) | present (8M)] f32
  char* w = (char*)d_ws;                   // 48 MiB used
  half_t* x16    = (half_t*)(w + (0ul  << 20));  // 8 MiB
  half_t* wattnT = (half_t*)(w + (8ul  << 20));  // 6 MiB
  half_t* wprojT = (half_t*)(w + (14ul << 20));  // 2 MiB
  half_t* q16    = (half_t*)(w + (16ul << 20));  // 8 MiB
  half_t* k16    = (half_t*)(w + (24ul << 20));  // 8 MiB
  half_t* vT16   = (half_t*)(w + (32ul << 20));  // 8 MiB
  half_t* a16    = (half_t*)(w + (40ul << 20));  // 8 MiB

  float* present = out + (long)Bb * Ss * Dd;

  cvt_kernel<<<16384, 256, 0, stream>>>(x, w_attn, w_proj, x16, wattnT, wprojT);
  qkv_gemm<<<dim3(32, 24), 256, 0, stream>>>(x16, wattnT, b_attn, q16, k16, vT16, present);
  attn_kernel<<<dim3(16, 64), 128, 0, stream>>>(q16, k16, vT16, ov, msk, a16);
  proj_gemm<<<dim3(32, 8), 256, 0, stream>>>(a16, wprojT, b_proj, out);
}
